// assignment_7344394076590
// MI455X (gfx1250) — matrix operand order repaired and checked by reading, not compiled, not run on hardware
//
#include <hip/hip_runtime.h>
#include <math.h>

typedef __attribute__((ext_vector_type(16))) _Float16 v16h;
typedef __attribute__((ext_vector_type(8)))  float    v8f;

#define NUM_CLASSES 80
#define CP 96          // K padded to 3 x 32 for wmma_f32_16x16x32_f16
#define NCAND 10

__device__ __forceinline__ float sigf(float x) { return 1.0f / (1.0f + __expf(-x)); }

// ---------------------------------------------------------------------------
// Kernel 1: per-anchor class-loss prep.
// diff[a,c] = log p - log1p(-p)  (f16, padded to CP), s_all[a] = sum_c log1p(-p)
// ---------------------------------------------------------------------------
__global__ void k_prep(const int* batch_idx, const float* cls_preds, const float* obj_preds,
                       int A, _Float16* diff, float* s_all_out)
{
    int a = blockIdx.x * blockDim.x + threadIdx.x;
    if (a >= A) return;
    int b = *batch_idx;
    const float* crow = cls_preds + ((size_t)b * A + a) * NUM_CLASSES;
    float so = sigf(obj_preds[(size_t)b * A + a]);
    float s_all = 0.f;
    _Float16* drow = diff + (size_t)a * CP;
#pragma unroll 4
    for (int c = 0; c < NUM_CLASSES; ++c) {
        float p = sqrtf(sigf(crow[c]) * so);
        p = fminf(fmaxf(p, 1e-7f), 1.0f - 1e-7f);
        float lp  = __logf(p);
        float l1p = log1pf(-p);
        s_all += l1p;
        drow[c] = (_Float16)(lp - l1p);
    }
    for (int c = NUM_CLASSES; c < CP; ++c) drow[c] = (_Float16)0.f;
    s_all_out[a] = s_all;
}

// ---------------------------------------------------------------------------
// Kernel 2: fg_any[a] = OR_g (in_box | in_ctr)
// ---------------------------------------------------------------------------
__global__ void k_fg(const float* gt, int G, const float* xs, const float* ys, const float* st,
                     int A, int* fg_any)
{
    int a = blockIdx.x * blockDim.x + threadIdx.x;
    if (a >= A) return;
    float s  = st[a];
    float xc = (xs[a] + 0.5f) * s;
    float yc = (ys[a] + 0.5f) * s;
    float r  = 2.5f * s;
    int fg = 0;
    for (int g = 0; g < G; ++g) {
        float gx = gt[4 * g + 0], gy = gt[4 * g + 1];
        float gw = gt[4 * g + 2], gh = gt[4 * g + 3];
        bool ib = (xc > gx - 0.5f * gw) && (gx + 0.5f * gw > xc) &&
                  (yc > gy - 0.5f * gh) && (gy + 0.5f * gh > yc);
        bool ic = (xc > gx - r) && (gx + r > xc) && (yc > gy - r) && (gy + r > yc);
        fg |= (ib || ic) ? 1 : 0;
    }
    fg_any[a] = fg;
}

// ---------------------------------------------------------------------------
// Kernel 3: zero per-anchor match state (ws is poisoned; must re-init per call)
// ---------------------------------------------------------------------------
__global__ void k_init(int* mcnt, int* mmin, int* nfg, int A)
{
    int a = blockIdx.x * blockDim.x + threadIdx.x;
    if (a < A) { mcnt[a] = 0; mmin[a] = 0x7fffffff; }
    if (a == 0) *nfg = 0;
}

// ---------------------------------------------------------------------------
// Kernel 4: pairwise cost + iou via WMMA one-hot matmul.
// Block = 256 thr = 8 waves; wave -> one 16(G) x 16(A) tile.
// E[g,a] = onehot(gt_class[g]) . diff[a,:]  computed by v_wmma_f32_16x16x32_f16.
// ---------------------------------------------------------------------------
__global__ void k_pair(const float* gt, const int* gcls, int G,
                       const float* pr, const float* xs, const float* ys, const float* st,
                       const _Float16* diff, const float* s_all, const int* fg_any,
                       int A, float* cost, float* ious)
{
    __shared__ float sgt[16][4];
    __shared__ int   scls[16];
    int g0 = blockIdx.y * 16;
    int t  = threadIdx.x;
    if (t < 64) { int g = g0 + (t >> 2); sgt[t >> 2][t & 3] = (g < G) ? gt[4 * g + (t & 3)] : 0.f; }
    if (t < 16) { int g = g0 + t;        scls[t] = (g < G) ? gcls[g] : -1; }
    __syncthreads();

    int wave = t >> 5;
    int lane = t & 31;
    int half = lane >> 4;
    int n    = lane & 15;
    int a0   = (blockIdx.x * 8 + wave) * 16;
    int a    = a0 + n;
    int ac   = min(a, A - 1);

    // per-anchor (per-lane column) data
    float px = pr[4 * ac + 0], py = pr[4 * ac + 1];
    float pw = pr[4 * ac + 2], ph = pr[4 * ac + 3];
    float sA  = st[ac];
    float xc  = (xs[ac] + 0.5f) * sA;
    float yc  = (ys[ac] + 0.5f) * sA;
    float rad = 2.5f * sA;
    float sall = s_all[ac];
    int   fga  = fg_any[ac];

    int cm = scls[n];                      // A-fragment row class (row m = lane&15)
    const _Float16* drow = diff + (size_t)ac * CP;

    v8f acc = {};
#pragma unroll
    for (int ks = 0; ks < 3; ++ks) {
        v16h afrag, bfrag;
        // 16-bit A layout: element j holds K = 16*half + j (contiguous, matches B)
#pragma unroll
        for (int j = 0; j < 16; ++j) {
            int kl = 16 * half + j;
            afrag[j] = (_Float16)((cm == 32 * ks + kl) ? 1.0f : 0.0f);
        }
        // 16-bit B layout: lanes 0-15 hold K=0..15, lanes 16-31 hold K=16..31 (contiguous)
        const _Float16* bsrc = drow + 32 * ks + 16 * half;
#pragma unroll
        for (int j = 0; j < 16; ++j) bfrag[j] = bsrc[j];
        acc = __builtin_amdgcn_wmma_f32_16x16x32_f16(false, afrag, false, bfrag,
                                                     (short)0, acc, false, false);
    }

    // C/D layout: element r -> (m = r + 8*half, n = lane&15)
#pragma unroll
    for (int r2 = 0; r2 < 8; ++r2) {
        int m = r2 + 8 * half;
        int g = g0 + m;
        if (g < G && a < A) {
            float gx = sgt[m][0], gy = sgt[m][1], gw = sgt[m][2], gh = sgt[m][3];
            float tlx = fmaxf(gx - 0.5f * gw, px - 0.5f * pw);
            float tly = fmaxf(gy - 0.5f * gh, py - 0.5f * ph);
            float brx = fminf(gx + 0.5f * gw, px + 0.5f * pw);
            float bry = fminf(gy + 0.5f * gh, py + 0.5f * ph);
            float iw = fmaxf(brx - tlx, 0.f), ih = fmaxf(bry - tly, 0.f);
            float inter = iw * ih;
            float iou   = inter / (gw * gh + pw * ph - inter);
            float iou_e = fga ? iou : 0.f;
            bool ib = (xc > gx - 0.5f * gw) && (gx + 0.5f * gw > xc) &&
                      (yc > gy - 0.5f * gh) && (gy + 0.5f * gh > yc);
            bool ic = (xc > gx - rad) && (gx + rad > xc) &&
                      (yc > gy - rad) && (gy + rad > yc);
            bool cand = ib && ic;
            float cls_loss = -(acc[r2] + sall);
            float cst = cls_loss - 3.0f * __logf(iou_e + 1e-8f)
                        + (cand ? 0.f : 100000.0f) + (fga ? 0.f : 1000000.0f);
            size_t o = (size_t)g * A + a;
            cost[o] = cst;
            ious[o] = iou_e;
        }
    }
}

// ---------------------------------------------------------------------------
// Kernel 5: per-GT top-10 (ious for dynamic-k, cost for selection). 1 block/GT.
// ---------------------------------------------------------------------------
__global__ __launch_bounds__(256)
void k_topk(const float* cost, const float* ious, int G, int A, int* mcnt, int* mmin)
{
    int g   = blockIdx.x;
    int tid = threadIdx.x;
    float ti[NCAND]; float tc[NCAND]; int tci[NCAND];
    for (int k = 0; k < NCAND; ++k) { ti[k] = -1e30f; tc[k] = 1e30f; tci[k] = -1; }
    const float* crow = cost + (size_t)g * A;
    const float* irow = ious + (size_t)g * A;
    for (int a = tid; a < A; a += 256) {
        float v = irow[a];
        if (v > ti[NCAND - 1]) {
            int k = NCAND - 1;
            while (k > 0 && v > ti[k - 1]) { ti[k] = ti[k - 1]; --k; }
            ti[k] = v;
        }
        float c = crow[a];
        if (c < tc[NCAND - 1]) {
            int k = NCAND - 1;
            while (k > 0 && c < tc[k - 1]) { tc[k] = tc[k - 1]; tci[k] = tci[k - 1]; --k; }
            tc[k] = c; tci[k] = a;
        }
    }
    __shared__ float shi[256][NCAND];
    __shared__ float shc[256][NCAND];
    __shared__ int   shci[256][NCAND];
    for (int k = 0; k < NCAND; ++k) { shi[tid][k] = ti[k]; shc[tid][k] = tc[k]; shci[tid][k] = tci[k]; }
    __syncthreads();
    for (int strd = 128; strd > 0; strd >>= 1) {
        if (tid < strd) {
            float mi[NCAND], mc[NCAND]; int mci[NCAND];
            int i = 0, j = 0;
            for (int k = 0; k < NCAND; ++k)
                mi[k] = (shi[tid][i] >= shi[tid + strd][j]) ? shi[tid][i++] : shi[tid + strd][j++];
            i = 0; j = 0;
            for (int k = 0; k < NCAND; ++k) {
                if (shc[tid][i] <= shc[tid + strd][j]) { mc[k] = shc[tid][i]; mci[k] = shci[tid][i]; ++i; }
                else                                   { mc[k] = shc[tid + strd][j]; mci[k] = shci[tid + strd][j]; ++j; }
            }
            for (int k = 0; k < NCAND; ++k) { shi[tid][k] = mi[k]; shc[tid][k] = mc[k]; shci[tid][k] = mci[k]; }
        }
        __syncthreads();
    }
    if (tid == 0) {
        float s = 0.f;
        for (int k = 0; k < NCAND; ++k) s += shi[0][k];
        int dk = (int)s;                              // trunc like int32 cast
        dk = max(1, min(NCAND, dk));
        for (int j2 = 0; j2 < dk; ++j2) {
            int aidx = shci[0][j2];
            if (aidx >= 0) {
                atomicAdd(&mcnt[aidx], 1);            // order-independent -> deterministic
                atomicMin(&mmin[aidx], g);
            }
        }
    }
}

// ---------------------------------------------------------------------------
// Kernel 6: per-anchor resolve + outputs.
// ---------------------------------------------------------------------------
__global__ void k_final(const float* cost, const float* ious, const int* gcls,
                        const int* mcnt, const int* mmin, int G, int A,
                        float* out, int* nfg)
{
    int a = blockIdx.x * blockDim.x + threadIdx.x;
    if (a >= A) return;
    float best = -1e38f; int bg = 0;                  // argmax(cost, axis=0), first max
    for (int g = 0; g < G; ++g) {
        float v = cost[(size_t)g * A + a];
        if (v > best) { best = v; bg = g; }
    }
    int cnt = mcnt[a];
    int matched; float pi, cls, fg;
    if (cnt == 0) { matched = -1; pi = 0.f; cls = -1.f; fg = 0.f; }
    else {
        int m = (cnt > 1) ? bg : mmin[a];
        matched = m;
        pi  = ious[(size_t)m * A + a];
        cls = (float)gcls[m];
        fg  = 1.f;
        atomicAdd(nfg, 1);
    }
    out[a]                 = cls;
    out[(size_t)A + a]     = fg;
    out[(size_t)2 * A + a] = pi;
    out[(size_t)3 * A + a] = (float)matched;
}

__global__ void k_nfg(const int* nfg, float* out, int A) { out[(size_t)4 * A] = (float)(*nfg); }

// ---------------------------------------------------------------------------
extern "C" void kernel_launch(void* const* d_in, const int* in_sizes, int n_in,
                              void* d_out, int out_size, void* d_ws, size_t ws_size,
                              hipStream_t stream)
{
    (void)n_in; (void)out_size; (void)ws_size;
    const int*   batch_idx = (const int*)d_in[0];
    int G = in_sizes[3] / 4;                 // gt_bboxes [G,4]
    int A = in_sizes[7];                     // x_shifts [1,A]
    const float* gt        = (const float*)d_in[3];
    const int*   gcls      = (const int*)d_in[4];
    const float* pr        = (const float*)d_in[5];
    const float* st        = (const float*)d_in[6];
    const float* xs        = (const float*)d_in[7];
    const float* ys        = (const float*)d_in[8];
    const float* cls_preds = (const float*)d_in[9];
    const float* obj_preds = (const float*)d_in[10];
    float* out = (float*)d_out;

    char* base = (char*)d_ws;
    size_t off = 0;
    auto up = [](size_t x) { return (x + 255) & ~(size_t)255; };
    _Float16* diff  = (_Float16*)(base + off); off = up(off + (size_t)A * CP * 2);
    float*    s_all = (float*)(base + off);    off = up(off + (size_t)A * 4);
    int*      fga   = (int*)(base + off);      off = up(off + (size_t)A * 4);
    float*    cost  = (float*)(base + off);    off = up(off + (size_t)G * A * 4);
    float*    ious  = (float*)(base + off);    off = up(off + (size_t)G * A * 4);
    int*      mcnt  = (int*)(base + off);      off = up(off + (size_t)A * 4);
    int*      mmin  = (int*)(base + off);      off = up(off + (size_t)A * 4);
    int*      nfg   = (int*)(base + off);      off = up(off + 4);

    const int thr = 256;
    int ablocks = (A + thr - 1) / thr;
    k_prep<<<ablocks, thr, 0, stream>>>(batch_idx, cls_preds, obj_preds, A, diff, s_all);
    k_fg  <<<ablocks, thr, 0, stream>>>(gt, G, xs, ys, st, A, fga);
    k_init<<<ablocks, thr, 0, stream>>>(mcnt, mmin, nfg, A);
    dim3 grid2((A + 127) / 128, (G + 15) / 16);
    k_pair<<<grid2, 256, 0, stream>>>(gt, gcls, G, pr, xs, ys, st, diff, s_all, fga, A, cost, ious);
    k_topk<<<G, 256, 0, stream>>>(cost, ious, G, A, mcnt, mmin);
    k_final<<<ablocks, thr, 0, stream>>>(cost, ious, gcls, mcnt, mmin, G, A, out, nfg);
    k_nfg<<<1, 1, 0, stream>>>(nfg, out, A);
}
